// InputMPNN_1571958031014
// MI455X (gfx1250) — matrix operand order repaired and checked by reading, not compiled, not run on hardware
//
#include <hip/hip_runtime.h>
#include <stdint.h>
#include <stddef.h>

// ---------------- problem constants (match reference) ----------------
#define TWO_PI_F 6.28318530717958647692f
#define HARD_CUT_F 1.73f
#define SOFT_CUT_F 1.73f
#define SOFT_W_F   0.2f
#define SLOPE_F    0.01f

#define BB 8
#define NN 256
#define CC 128          // C
#define LWID 256        // LW = 2*COUT
#define KA 48           // padded A' rows: 32 basis + 1 cut + 15 zero
#define LDA 264         // LDS row stride (f16) for A'  (264*2 = 528 = 16*33 B)

typedef _Float16 v8h  __attribute__((ext_vector_type(8)));
typedef _Float16 v16h __attribute__((ext_vector_type(16)));
typedef float    v8f  __attribute__((ext_vector_type(8)));

union V16U { v16h v; v8h h[2]; };

// A-fragment (16x32 f16): lane m=lane%16 holds 16 contiguous K values
// K = kb .. kb+15 (kb = 0 for lanes 0-15, 16 for 16-31), matching B.
static __device__ inline v16h load_afrag(const _Float16* lo) {
  V16U u;
  u.h[0] = *(const v8h*)lo;          // K = kb .. kb+7
  u.h[1] = *(const v8h*)(lo + 8);    // K = kb+8 .. kb+15
  return u.v;
}

// ---------------- kernel 0: precision/layout conversion ----------------
// Ft[(b*CC+c)*NN + x] = f16(features[(b*NN+x)*CC + c])   (transposed, x-contig)
// W1t[n*256+k] = f16(W1[k*256+n]) ; W2t likewise (B^T layout, k-contig)
__global__ void k0_convert(const float* __restrict__ feat,
                           const float* __restrict__ W1,
                           const float* __restrict__ W2,
                           _Float16* __restrict__ Ft,
                           _Float16* __restrict__ W1t,
                           _Float16* __restrict__ W2t) {
  int idx = blockIdx.x * blockDim.x + threadIdx.x;
  if (idx < BB * CC * NN) {
    int b = idx / (CC * NN);
    int c = (idx / NN) % CC;
    int x = idx % NN;
    Ft[idx] = (_Float16)feat[((size_t)b * NN + x) * CC + c];
  }
  int i = idx - BB * CC * NN;
  if (i >= 0 && i < 256 * 256) {
    int n = i / 256, k = i % 256;
    W1t[i] = (_Float16)W1[k * 256 + n];
  }
  int j = idx - BB * CC * NN - 256 * 256;
  if (j >= 0 && j < 256 * 256) {
    int n = j / 256, k = j % 256;
    W2t[j] = (_Float16)W2[k * 256 + n];
  }
}

// ---------------- kernel 1: fused radial basis + cutoff + aggregation ----------------
// One workgroup per (b,a). Phase 1: build A'[k][x] in LDS (f16).
// Phase 2: M2 = A'^T(48x256) @ F(256x128) via v_wmma_f32_16x16x32_f16,
// epilogue folds rad_W / rad_b: fmp[c] = sum_k W'[k,c]*M2[k,c].
__global__ __launch_bounds__(256) void k1_fmp(
    const float* __restrict__ norms, const uint8_t* __restrict__ emask,
    const _Float16* __restrict__ Ft, const float* __restrict__ radW,
    const float* __restrict__ radB, float* __restrict__ fmp) {
  __shared__ __align__(32) _Float16 As[KA * LDA];
  const int row = blockIdx.x;          // b*NN + a
  const int b   = row >> 8;
  const int tid = threadIdx.x;

  // ---- phase 1: per-edge basis (one x per thread) ----
  {
    const int x = tid;
    const size_t e = (size_t)row * NN + x;
    const float r = norms[e];
    const bool em = emask[e] != 0;
    const bool mb = em && (r > 0.f);
    const float inv = mb ? (1.f / r) : 0.f;
    float pw[4];
    pw[0] = mb ? 1.f : 0.f;
    pw[1] = inv;
    pw[2] = inv * inv;
    pw[3] = pw[2] * inv;
    float cutv = 0.f;
    if (em && (r < HARD_CUT_F))
      cutv = 1.f / (1.f + __expf((r - SOFT_CUT_F) * (1.f / SOFT_W_F)));
    float tg[8];
#pragma unroll
    for (int t = 0; t < 4; ++t) {
      const float ang = TWO_PI_F * (float)t * r;
      tg[t]     = mb ? __sinf(ang) : 0.f;   // phase 0
      tg[t + 4] = mb ? __cosf(ang) : 0.f;   // phase pi/2
    }
#pragma unroll
    for (int t = 0; t < 8; ++t)
#pragma unroll
      for (int q = 0; q < 4; ++q)
        As[(t * 4 + q) * LDA + x] = (_Float16)(cutv * tg[t] * pw[q]);
    As[32 * LDA + x] = (_Float16)cutv;      // carries rad_b * cut term
#pragma unroll
    for (int k = 33; k < KA; ++k) As[k * LDA + x] = (_Float16)0.f;
  }
  __syncthreads();

  // ---- phase 2: WMMA, one 16-channel slab per wave (8 waves * 16 = 128 = C) ----
  const int wave = tid >> 5, lane = tid & 31;
  const int n = lane & 15, half = lane >> 4;
  const int c = wave * 16 + n;
  const _Float16* Fb = Ft + ((size_t)b * CC + c) * NN;  // this lane's B column, x-contig
  const int kbA = half * 16;   // A-fragment K sub-offset
  const int kbB = half * 16;   // B-fragment K sub-offset

  v8f acc0 = {}, acc1 = {}, acc2 = {};
#pragma unroll
  for (int kc = 0; kc < 8; ++kc) {                  // K = 256 over x, 32/step
    const v16h bf = *(const v16h*)(Fb + kc * 32 + kbB);
    const _Float16* a0 = &As[(0 * 16 + n) * LDA + kc * 32 + kbA];
    const _Float16* a1 = &As[(1 * 16 + n) * LDA + kc * 32 + kbA];
    const _Float16* a2 = &As[(2 * 16 + n) * LDA + kc * 32 + kbA];
    acc0 = __builtin_amdgcn_wmma_f32_16x16x32_f16(false, load_afrag(a0), false, bf,
                                                  (short)0, acc0, false, false);
    acc1 = __builtin_amdgcn_wmma_f32_16x16x32_f16(false, load_afrag(a1), false, bf,
                                                  (short)0, acc1, false, false);
    acc2 = __builtin_amdgcn_wmma_f32_16x16x32_f16(false, load_afrag(a2), false, bf,
                                                  (short)0, acc2, false, false);
  }

  // ---- epilogue: fmp[c] = sum_k W'[k,c] * M2[k,c] ----
  // D layout: lane holds (M = v + 8*half, N = lane%16) in acc[v].
  float s = 0.f;
#pragma unroll
  for (int v = 0; v < 8; ++v) {
    const int k0 = 0 * 16 + half * 8 + v;
    const int k1 = 1 * 16 + half * 8 + v;
    const int k2 = 2 * 16 + half * 8 + v;
    s += radW[k0 * CC + c] * acc0[v];
    s += radW[k1 * CC + c] * acc1[v];
    if (k2 == 32) s += radB[c] * acc2[v];   // cut row; rows 33..47 are zero
  }
  s += __shfl_xor(s, 16, 32);               // combine M halves of the column
  if (half == 0) fmp[(size_t)row * CC + c] = s;
}

// ---------------- kernel 2a: X = concat(fmp, features) -> f16 ----------------
__global__ void k2a_buildX(const float* __restrict__ fmp,
                           const float* __restrict__ feat,
                           _Float16* __restrict__ X16) {
  const int idx = blockIdx.x * blockDim.x + threadIdx.x; // 2048*256
  const int r = idx >> 8, k = idx & 255;
  const float v = (k < CC) ? fmp[(size_t)r * CC + k]
                           : feat[(size_t)r * CC + (k - CC)];
  X16[idx] = (_Float16)v;
}

// ---------------- kernel 2b: H = leakyrelu(X @ W1 + b1) ----------------
// One 16x16 tile per wave; M=2048, N=256, K=256 (8 WMMA steps).
__global__ __launch_bounds__(256) void k2_gemm1(
    const _Float16* __restrict__ X16, const _Float16* __restrict__ W1t,
    const float* __restrict__ b1, _Float16* __restrict__ H16) {
  const int wave = threadIdx.x >> 5, lane = threadIdx.x & 31;
  const int tile = blockIdx.x * 8 + wave;    // 2048 tiles = 128 Mt * 16 Nt
  const int mt = tile >> 4, nt = tile & 15;
  const int n = lane & 15, half = lane >> 4;
  const _Float16* Arow = X16 + ((size_t)(mt * 16 + n)) * 256 + half * 16;
  const _Float16* Brow = W1t + ((size_t)(nt * 16 + n)) * 256 + half * 16;
  v8f acc = {};
#pragma unroll
  for (int kc = 0; kc < 8; ++kc) {
    const v16h af = load_afrag(Arow + kc * 32);
    const v16h bf = *(const v16h*)(Brow + kc * 32);
    acc = __builtin_amdgcn_wmma_f32_16x16x32_f16(false, af, false, bf,
                                                 (short)0, acc, false, false);
  }
  const int col = nt * 16 + n;
  const float bias = b1[col];
#pragma unroll
  for (int v = 0; v < 8; ++v) {
    const int m = mt * 16 + half * 8 + v;
    float h = acc[v] + bias;
    h = (h > 0.f) ? h : SLOPE_F * h;
    H16[(size_t)m * 256 + col] = (_Float16)h;
  }
}

// ---------------- kernel 2c: Y = (H @ W2 + b2) * atom_mask -> f32 out ----------------
__global__ __launch_bounds__(256) void k2_gemm2(
    const _Float16* __restrict__ H16, const _Float16* __restrict__ W2t,
    const float* __restrict__ b2, const uint8_t* __restrict__ amask,
    float* __restrict__ out) {
  const int wave = threadIdx.x >> 5, lane = threadIdx.x & 31;
  const int tile = blockIdx.x * 8 + wave;
  const int mt = tile >> 4, nt = tile & 15;
  const int n = lane & 15, half = lane >> 4;
  const _Float16* Arow = H16 + ((size_t)(mt * 16 + n)) * 256 + half * 16;
  const _Float16* Brow = W2t + ((size_t)(nt * 16 + n)) * 256 + half * 16;
  v8f acc = {};
#pragma unroll
  for (int kc = 0; kc < 8; ++kc) {
    const v16h af = load_afrag(Arow + kc * 32);
    const v16h bf = *(const v16h*)(Brow + kc * 32);
    acc = __builtin_amdgcn_wmma_f32_16x16x32_f16(false, af, false, bf,
                                                 (short)0, acc, false, false);
  }
  const int col = nt * 16 + n;
  const float bias = b2[col];
#pragma unroll
  for (int v = 0; v < 8; ++v) {
    const int m = mt * 16 + half * 8 + v;
    const float y = acc[v] + bias;
    out[(size_t)m * 256 + col] = amask[m] ? y : 0.f;
  }
}

// ---------------- host-side launcher ----------------
extern "C" void kernel_launch(void* const* d_in, const int* in_sizes, int n_in,
                              void* d_out, int out_size, void* d_ws, size_t ws_size,
                              hipStream_t stream) {
  (void)in_sizes; (void)n_in; (void)out_size; (void)ws_size;
  const float*   feat  = (const float*)d_in[0];
  const uint8_t* amask = (const uint8_t*)d_in[1];
  /* d_in[2] edge_features: unused by the module */
  const uint8_t* emask = (const uint8_t*)d_in[3];
  const float*   norms = (const float*)d_in[4];
  const float*   radW  = (const float*)d_in[5];
  const float*   radB  = (const float*)d_in[6];
  const float*   W1    = (const float*)d_in[7];
  const float*   b1    = (const float*)d_in[8];
  const float*   W2    = (const float*)d_in[9];
  const float*   b2    = (const float*)d_in[10];

  // workspace layout (bytes), total ~3.93 MB
  char* ws = (char*)d_ws;
  _Float16* Ft  = (_Float16*)(ws + 0);         // 512 KB  (B*C*N f16, transposed feats)
  _Float16* X16 = (_Float16*)(ws + 524288);    // 1 MB    (2048x256 f16)
  _Float16* H16 = (_Float16*)(ws + 1572864);   // 1 MB    (2048x256 f16)
  _Float16* W1t = (_Float16*)(ws + 2621440);   // 128 KB
  _Float16* W2t = (_Float16*)(ws + 2752512);   // 128 KB
  float*    fmp = (float*)   (ws + 2883584);   // 1 MB    (2048x128 f32)

  k0_convert<<<1536, 256, 0, stream>>>(feat, W1, W2, Ft, W1t, W2t);
  k1_fmp    <<<BB * NN, 256, 0, stream>>>(norms, emask, Ft, radW, radB, fmp);
  k2a_buildX<<<2048, 256, 0, stream>>>(fmp, feat, X16);
  k2_gemm1  <<<256, 256, 0, stream>>>(X16, W1t, b1, H16);
  k2_gemm2  <<<256, 256, 0, stream>>>(H16, W2t, b2, amask, (float*)d_out);
}
